// PWCFlow_20280835572148
// MI455X (gfx1250) — matrix operand order repaired and checked by reading, not compiled, not run on hardware
//
#include <hip/hip_runtime.h>

typedef __bf16 bf16_t;
typedef __attribute__((ext_vector_type(16))) __bf16 v16bf;
typedef __attribute__((ext_vector_type(8)))  __bf16 v8bf;
typedef __attribute__((ext_vector_type(8)))  float v8f;

__device__ __forceinline__ float lrelu_f(float v) { return v >= 0.0f ? v : 0.1f * v; }

// ---------------------------------------------------------------------------
// pack: fp32 NCHW slice -> bf16 pixel-major channel-contiguous, channels
// padded to Cp (multiple of 32) with zeros; one extra zeroed guard pixel at
// index B*H*W used by the conv for border taps.
// ---------------------------------------------------------------------------
__global__ void pack_kernel(const float* __restrict__ src, bf16_t* __restrict__ dst,
                            int B, int CinTot, int cOff, int C, int H, int W, int Cp)
{
    const long idx   = (long)blockIdx.x * blockDim.x + threadIdx.x;
    const long npix  = (long)B * H * W;
    const long total = (npix + 1) * Cp;
    if (idx >= total) return;
    const int  c   = (int)(idx % Cp);
    const long pix = idx / Cp;
    float v = 0.0f;
    if (pix < npix && c < C) {
        const int w = (int)(pix % W);
        long t = pix / W;
        const int h = (int)(t % H);
        const int b = (int)(t / H);
        v = src[(((long)b * CinTot + cOff + c) * H + h) * W + w];
    }
    dst[idx] = (bf16_t)v;
}

// ---------------------------------------------------------------------------
// 3x3 conv as 9 tap-GEMMs on the CDNA5 bf16 WMMA path, 2x2 register tiling.
// Input is pre-packed bf16 (pixel-major, Cp-padded channels, zero guard pixel)
// so each lane's B fragment is ONE contiguous 32B load; border taps redirect
// the pixel index to the guard pixel (single select, no per-element masks).
// Weights for all 9 taps of the 32x32 chunk staged once per chunk in LDS;
// out-of-range K contributes zero via the A side.
// Fragment packing used here (wave32; contiguous k on A and B):
//   A 16x32 bf16 : m=lane&15; one run [khi*16..+15] of the row (K = kb+16*khi+e).
//   B 32x16 bf16 : N = lane&15; lanes 0-15 K = kb+e, lanes 16-31 K = kb+16+e.
//   C 16x16 f32  : VGPR v -> M = v + 8*(lane>=16), N = lane&15.
// ---------------------------------------------------------------------------
__global__ __launch_bounds__(128) void conv3x3_wmma_kernel(
    const bf16_t* __restrict__ xp, const float* __restrict__ wgt,
    const float* __restrict__ bias, float* __restrict__ y,
    int B, int Cin, int Cp, int H, int W,
    int Cout, int Ho, int Wo, int stride, int dil, int relu)
{
    __shared__ bf16_t sW[9 * 1024];         // [tap][co 32][ci 32]
    const int tid  = threadIdx.x;
    const int lane = tid & 31;
    const int wave = tid >> 5;
    const int Ntot = B * Ho * Wo;
    const int co0  = blockIdx.y * 32;
    const int n0   = (blockIdx.x * 4 + wave) * 32;   // two 16-pixel tiles/wave
    const int khi  = lane >> 4;
    const int m    = lane & 15;
    const long gpix = (long)B * H * W;      // guard (zero) pixel index

    int bb[2], oh[2], ow[2];
    bool nv[2];
#pragma unroll
    for (int t = 0; t < 2; ++t) {
        const int nl = n0 + t * 16 + m;
        nv[t] = (nl < Ntot);
        const int nlc = nv[t] ? nl : 0;
        const int b_ = nlc / (Ho * Wo);
        const int r  = nlc - b_ * Ho * Wo;
        bb[t] = b_;
        oh[t] = r / Wo;
        ow[t] = r - oh[t] * Wo;
    }

    __builtin_prefetch(xp + (((long)bb[0] * H + oh[0] * stride) * W + ow[0] * stride) * Cp,
                       0, 1);               // global_prefetch: warm L2/WGP$

    v8f acc00 = {0.f, 0.f, 0.f, 0.f, 0.f, 0.f, 0.f, 0.f};  // co-sub 0, tile 0
    v8f acc01 = {0.f, 0.f, 0.f, 0.f, 0.f, 0.f, 0.f, 0.f};  // co-sub 0, tile 1
    v8f acc10 = {0.f, 0.f, 0.f, 0.f, 0.f, 0.f, 0.f, 0.f};  // co-sub 1, tile 0
    v8f acc11 = {0.f, 0.f, 0.f, 0.f, 0.f, 0.f, 0.f, 0.f};  // co-sub 1, tile 1

    for (int kb = 0; kb < Cin; kb += 32) {
        __syncthreads();
        // ---- stage 9 x 32(co) x 32(ci) weight tile as bf16 in LDS ----
        unsigned* sWu = (unsigned*)sW;
#pragma unroll
        for (int p = 0; p < 4; ++p) {
            const int sp = tid + p * 128;   // pair id 0..511
            const int s  = sp * 2;          // even element id
            const int mm = s >> 5, kk = s & 31;
            const int k0 = kb + kk, k1 = k0 + 1;
            const int co = co0 + mm;
            const int coc = co < Cout ? co : Cout - 1;
            const int k0c = k0 < Cin ? k0 : Cin - 1;
            const int k1c = k1 < Cin ? k1 : Cin - 1;
            const bool ok0 = (k0 < Cin) && (co < Cout);
            const bool ok1 = (k1 < Cin) && (co < Cout);
            const float* wp0 = wgt + ((long)coc * Cin + k0c) * 9;
            const float* wp1 = wgt + ((long)coc * Cin + k1c) * 9;
#pragma unroll
            for (int tap = 0; tap < 9; ++tap) {
                float f0 = wp0[tap]; if (!ok0) f0 = 0.f;
                float f1 = wp1[tap]; if (!ok1) f1 = 0.f;
                union { bf16_t h[2]; unsigned u; } pk;
                pk.h[0] = (bf16_t)f0;
                pk.h[1] = (bf16_t)f1;
                sWu[(tap * 1024 + s) >> 1] = pk.u;
            }
        }
        __syncthreads();

#pragma unroll
        for (int tap = 0; tap < 9; ++tap) {
            const int kh = tap / 3;          // constant-folded
            const int kw = tap - kh * 3;

            // A fragments: each lane's row is two contiguous 16B runs in LDS
            const bf16_t* tapW = sW + tap * 1024;
            const v8bf* row0 = (const v8bf*)(tapW + m * 32);
            const v8bf* row1 = (const v8bf*)(tapW + (m + 16) * 32);
            const v16bf afrag0 = __builtin_shufflevector(
                row0[2 * khi], row0[2 * khi + 1],
                0, 1, 2, 3, 4, 5, 6, 7, 8, 9, 10, 11, 12, 13, 14, 15);
            const v16bf afrag1 = __builtin_shufflevector(
                row1[2 * khi], row1[2 * khi + 1],
                0, 1, 2, 3, 4, 5, 6, 7, 8, 9, 10, 11, 12, 13, 14, 15);

            // B fragments: one contiguous 32B load per tile; invalid taps
            // redirect to the zeroed guard pixel (no per-element masking).
            long pix[2];
#pragma unroll
            for (int t = 0; t < 2; ++t) {
                const int ih = oh[t] * stride + (kh - 1) * dil;
                const int iw = ow[t] * stride + (kw - 1) * dil;
                const bool tv = nv[t] && (ih >= 0) && (ih < H) && (iw >= 0) && (iw < W);
                pix[t] = tv ? (((long)bb[t] * H + ih) * W + iw) : gpix;
            }
            const v16bf bf0 = *(const v16bf*)(xp + pix[0] * Cp + kb + khi * 16);
            const v16bf bf1 = *(const v16bf*)(xp + pix[1] * Cp + kb + khi * 16);

            acc00 = __builtin_amdgcn_wmma_f32_16x16x32_bf16(
                false, afrag0, false, bf0, (short)0, acc00, false, false);
            acc01 = __builtin_amdgcn_wmma_f32_16x16x32_bf16(
                false, afrag0, false, bf1, (short)0, acc01, false, false);
            acc10 = __builtin_amdgcn_wmma_f32_16x16x32_bf16(
                false, afrag1, false, bf0, (short)0, acc10, false, false);
            acc11 = __builtin_amdgcn_wmma_f32_16x16x32_bf16(
                false, afrag1, false, bf1, (short)0, acc11, false, false);
        }
    }

#pragma unroll
    for (int v = 0; v < 8; ++v) {
        const int coA = co0 + v + khi * 8;           // co-subtile 0
        const int coB = coA + 16;                    // co-subtile 1
        if (coA < Cout) {
            const float bv = bias[coA];
            if (nv[0]) {
                float val = acc00[v] + bv;
                if (relu) val = lrelu_f(val);
                y[(((long)bb[0] * Cout + coA) * Ho + oh[0]) * Wo + ow[0]] = val;
            }
            if (nv[1]) {
                float val = acc01[v] + bv;
                if (relu) val = lrelu_f(val);
                y[(((long)bb[1] * Cout + coA) * Ho + oh[1]) * Wo + ow[1]] = val;
            }
        }
        if (coB < Cout) {
            const float bv = bias[coB];
            if (nv[0]) {
                float val = acc10[v] + bv;
                if (relu) val = lrelu_f(val);
                y[(((long)bb[0] * Cout + coB) * Ho + oh[0]) * Wo + ow[0]] = val;
            }
            if (nv[1]) {
                float val = acc11[v] + bv;
                if (relu) val = lrelu_f(val);
                y[(((long)bb[1] * Cout + coB) * Ho + oh[1]) * Wo + ow[1]] = val;
            }
        }
    }
}

// ---------------------------------------------------------------------------
// correlation: out[b, dy*9+dx, h, w] = mean_c x1[b,c,h,w]*x2[b,c,h+dy-4,w+dx-4]
// written at channel offset cOff into a buffer with Cd total channels.
// ---------------------------------------------------------------------------
__global__ void corr_kernel(const float* __restrict__ x1, const float* __restrict__ x2,
                            float* __restrict__ out, int B, int C, int H, int W,
                            int Cd, int cOff, int relu)
{
    const long idx   = (long)blockIdx.x * blockDim.x + threadIdx.x;
    const long total = (long)B * 81 * H * W;
    if (idx >= total) return;
    int w = (int)(idx % W);
    long t = idx / W;
    int h = (int)(t % H); t /= H;
    int o = (int)(t % 81);
    int b = (int)(t / 81);
    const int dy = o / 9 - 4, dx = o % 9 - 4;
    const int h2 = h + dy, w2 = w + dx;
    float s = 0.0f;
    if (h2 >= 0 && h2 < H && w2 >= 0 && w2 < W) {
        for (int c = 0; c < C; ++c)
            s += x1[(((size_t)b * C + c) * H + h) * W + w] *
                 x2[(((size_t)b * C + c) * H + h2) * W + w2];
    }
    float v = s / (float)C;
    if (relu) v = lrelu_f(v);
    out[(((size_t)b * Cd + cOff + o) * H + h) * W + w] = v;
}

// ---------------------------------------------------------------------------
// bilinear backward warp (zero outside)
// ---------------------------------------------------------------------------
__global__ void warp_kernel(const float* __restrict__ x, const float* __restrict__ fl,
                            float* __restrict__ out, int B, int C, int H, int W)
{
    const long idx   = (long)blockIdx.x * blockDim.x + threadIdx.x;
    const long total = (long)B * C * H * W;
    if (idx >= total) return;
    int w = (int)(idx % W);
    long t = idx / W;
    int h = (int)(t % H); t /= H;
    int c = (int)(t % C);
    int b = (int)(t / C);
    const size_t hw = (size_t)H * W;
    const float px = (float)w + fl[((size_t)b * 2 + 0) * hw + (size_t)h * W + w];
    const float py = (float)h + fl[((size_t)b * 2 + 1) * hw + (size_t)h * W + w];
    const float x0f = floorf(px), y0f = floorf(py);
    const float wx = px - x0f, wy = py - y0f;
    const float* base = x + ((size_t)b * C + c) * hw;
    auto g = [&](float yc, float xc) -> float {
        const bool valid = (xc >= 0.f) && (xc <= (float)(W - 1)) &&
                           (yc >= 0.f) && (yc <= (float)(H - 1));
        const int xi = (int)fminf(fmaxf(xc, 0.f), (float)(W - 1));
        const int yi = (int)fminf(fmaxf(yc, 0.f), (float)(H - 1));
        const float v = base[(size_t)yi * W + xi];
        return valid ? v : 0.f;
    };
    out[idx] = g(y0f, x0f)       * (1.f - wx) * (1.f - wy)
             + g(y0f, x0f + 1.f) * wx         * (1.f - wy)
             + g(y0f + 1.f, x0f) * (1.f - wx) * wy
             + g(y0f + 1.f, x0f + 1.f) * wx   * wy;
}

// ---------------------------------------------------------------------------
// align-corners bilinear resize with fused value scale (resize(s*x)=s*resize(x))
// ---------------------------------------------------------------------------
__global__ void resize_kernel(const float* __restrict__ x, float* __restrict__ out,
                              int B, int C, int H, int W, int OH, int OW, float scale)
{
    const long idx   = (long)blockIdx.x * blockDim.x + threadIdx.x;
    const long total = (long)B * C * OH * OW;
    if (idx >= total) return;
    int ow = (int)(idx % OW);
    long t = idx / OW;
    int oh = (int)(t % OH); t /= OH;
    int c = (int)(t % C);
    int b = (int)(t / C);
    const float ys = (OH > 1) ? (float)oh * (float)(H - 1) / (float)(OH - 1) : 0.f;
    const float xs = (OW > 1) ? (float)ow * (float)(W - 1) / (float)(OW - 1) : 0.f;
    int y0 = (int)floorf(ys); int y1 = (y0 + 1 < H - 1) ? y0 + 1 : H - 1;
    int x0 = (int)floorf(xs); int x1 = (x0 + 1 < W - 1) ? x0 + 1 : W - 1;
    const float wy = ys - (float)y0, wx = xs - (float)x0;
    const float* base = x + ((size_t)b * C + c) * H * W;
    const float v = (base[(size_t)y0 * W + x0] * (1.f - wy) + base[(size_t)y1 * W + x0] * wy) * (1.f - wx)
                  + (base[(size_t)y0 * W + x1] * (1.f - wy) + base[(size_t)y1 * W + x1] * wy) * wx;
    out[idx] = v * scale;
}

__global__ void copy_ch_kernel(const float* __restrict__ src, float* __restrict__ dst,
                               int B, int C, int H, int W, int Cd, int cOff)
{
    const long idx   = (long)blockIdx.x * blockDim.x + threadIdx.x;
    const long total = (long)B * C * H * W;
    if (idx >= total) return;
    int w = (int)(idx % W);
    long t = idx / W;
    int h = (int)(t % H); t /= H;
    int c = (int)(t % C);
    int b = (int)(t / C);
    dst[(((size_t)b * Cd + cOff + c) * H + h) * W + w] = src[idx];
}

__global__ void add_kernel(float* __restrict__ dst, const float* __restrict__ src, long n)
{
    const long idx = (long)blockIdx.x * blockDim.x + threadIdx.x;
    if (idx < n) dst[idx] += src[idx];
}

// ---------------------------------------------------------------------------
extern "C" void kernel_launch(void* const* d_in, const int* in_sizes, int n_in,
                              void* d_out, int out_size, void* d_ws, size_t ws_size,
                              hipStream_t stream)
{
    (void)in_sizes; (void)n_in; (void)out_size; (void)ws_size;
    const int B = 4;
    const int Hl[7] = {384, 192, 96, 48, 24, 12, 6};
    const int Wl[7] = {832, 416, 208, 104, 52, 26, 13};
    const int CHS[6] = {16, 32, 64, 96, 128, 192};   // LV_CHS[1:]

    // ---- parameter unpacking (flattened leaves in dict order) ----
    const float* X = (const float*)d_in[0];
    int pi = 1;
    const float *fpe_w1[6], *fpe_b1[6], *fpe_w2[6], *fpe_b2[6];
    for (int l = 0; l < 6; ++l) {
        fpe_w1[l] = (const float*)d_in[pi++]; fpe_b1[l] = (const float*)d_in[pi++];
        fpe_w2[l] = (const float*)d_in[pi++]; fpe_b2[l] = (const float*)d_in[pi++];
    }
    const float *est_w[5][6], *est_b[5][6];
    for (int i = 0; i < 5; ++i)
        for (int j = 0; j < 6; ++j) {
            est_w[i][j] = (const float*)d_in[pi++];
            est_b[i][j] = (const float*)d_in[pi++];
        }
    const float *ctx_w[7], *ctx_b[7];
    for (int j = 0; j < 7; ++j) {
        ctx_w[j] = (const float*)d_in[pi++];
        ctx_b[j] = (const float*)d_in[pi++];
    }

    // ---- workspace bump allocator (graph-capture safe) ----
    char* wsp = (char*)d_ws;
    size_t woff = 0;
    auto allocB = [&](size_t bytes) -> void* {
        bytes = (bytes + 255) & ~(size_t)255;
        void* p = (void*)(wsp + woff);
        woff += bytes;
        return p;
    };
    auto alloc = [&](size_t nfloats) -> float* {
        return (float*)allocB(nfloats * sizeof(float));
    };

    auto blocks1 = [](long n) { return dim3((unsigned)((n + 255) / 256)); };

    // pack buffer sized for the largest conv input: level-0 image, Cp=32
    bf16_t* packBuf = (bf16_t*)allocB(((size_t)B * 384 * 832 + 1) * 32 * sizeof(bf16_t));

    auto conv = [&](const float* xin, int CinTot, int cOff, int Cin, int H, int W,
                    const float* w, const float* b, float* y, int Cout,
                    int Ho, int Wo, int stride, int dil, int relu) {
        const int Cp = (Cin + 31) & ~31;
        const long npk = ((long)B * H * W + 1) * Cp;
        pack_kernel<<<blocks1(npk), 256, 0, stream>>>(
            xin, packBuf, B, CinTot, cOff, Cin, H, W, Cp);
        dim3 grid((unsigned)(((long)B * Ho * Wo + 127) / 128),
                  (unsigned)((Cout + 31) / 32));
        conv3x3_wmma_kernel<<<grid, 128, 0, stream>>>(
            packBuf, w, b, y, B, Cin, Cp, H, W, Cout, Ho, Wo, stride, dil, relu);
    };

    // ---- feature pyramids for both images ----
    float* tmp = alloc((size_t)B * 16 * 192 * 416);   // largest intermediate
    float* feats[2][6];
    for (int img = 0; img < 2; ++img) {
        const float* cur = X;
        int curCt = 6, curOff = img * 3, curC = 3;
        for (int l = 0; l < 6; ++l) {
            const int Co = CHS[l], Ho = Hl[l + 1], Wo = Wl[l + 1];
            conv(cur, curCt, curOff, curC, Hl[l], Wl[l],
                 fpe_w1[l], fpe_b1[l], tmp, Co, Ho, Wo, 2, 1, 1);
            feats[img][l] = alloc((size_t)B * Co * Ho * Wo);
            conv(tmp, Co, 0, Co, Ho, Wo,
                 fpe_w2[l], fpe_b2[l], feats[img][l], Co, Ho, Wo, 1, 1, 1);
            cur = feats[img][l]; curCt = Co; curOff = 0; curC = Co;
        }
    }

    // ---- shared flow-loop scratch (sized for finest level: 96x208) ----
    const size_t maxPix = (size_t)B * 96 * 208;
    float* pingA  = alloc(maxPix * 128);
    float* pingB  = alloc(maxPix * 128);
    float* est_in = alloc(maxPix * 115);
    float* x2w    = alloc((size_t)B * 32 * 96 * 208);
    float* upf    = alloc(maxPix * 2);

    const int estCh[5] = {192, 128, 96, 64, 32};
    const int hid[5]   = {128, 128, 96, 64, 32};
    float* flows[5]; int fH[5], fW[5];
    float* flow = nullptr; int ph = 0, pw = 0;

    for (int i = 0; i < 5; ++i) {
        const int lvl = 5 - i;                 // p[i] = feats[5-i]
        const float* x1 = feats[0][lvl];
        const float* x2 = feats[1][lvl];
        const int C = estCh[i], h = Hl[lvl + 1], w = Wl[lvl + 1];
        int cin;
        if (i == 0) {
            cin = 81;
            corr_kernel<<<blocks1((long)B * 81 * h * w), 256, 0, stream>>>(
                x1, x2, est_in, B, C, h, w, cin, 0, 0);
        } else {
            cin = 81 + C + 2;
            resize_kernel<<<blocks1((long)B * 2 * h * w), 256, 0, stream>>>(
                flow, upf, B, 2, ph, pw, h, w, 2.0f);
            warp_kernel<<<blocks1((long)B * C * h * w), 256, 0, stream>>>(
                x2, upf, x2w, B, C, h, w);
            corr_kernel<<<blocks1((long)B * 81 * h * w), 256, 0, stream>>>(
                x1, x2w, est_in, B, C, h, w, cin, 0, 1);
            copy_ch_kernel<<<blocks1((long)B * C * h * w), 256, 0, stream>>>(
                x1, est_in, B, C, h, w, cin, 81);
            copy_ch_kernel<<<blocks1((long)B * 2 * h * w), 256, 0, stream>>>(
                upf, est_in, B, 2, h, w, cin, 81 + C);
        }

        // flow estimator stack
        const float* in = est_in; int ic = cin;
        float* outs[5] = {pingA, pingB, pingA, pingB, pingA};
        for (int j = 0; j < 5; ++j) {
            conv(in, ic, 0, ic, h, w, est_w[i][j], est_b[i][j], outs[j], hid[j], h, w, 1, 1, 1);
            in = outs[j]; ic = hid[j];
        }
        float* feat = pingA;                   // 32-ch hidden features
        flows[i] = alloc((size_t)B * 2 * h * w);
        conv(feat, 32, 0, 32, h, w, est_w[i][5], est_b[i][5], flows[i], 2, h, w, 1, 1, 0);
        if (i > 0)
            add_kernel<<<blocks1((long)B * 2 * h * w), 256, 0, stream>>>(
                flows[i], upf, (long)B * 2 * h * w);

        if (i == 4) {
            // context network on concat(flow, feat)
            float* ctx_in = est_in;            // reuse (34 ch)
            copy_ch_kernel<<<blocks1((long)B * 2 * h * w), 256, 0, stream>>>(
                flows[4], ctx_in, B, 2, h, w, 34, 0);
            copy_ch_kernel<<<blocks1((long)B * 32 * h * w), 256, 0, stream>>>(
                feat, ctx_in, B, 32, h, w, 34, 2);
            const int dils[6] = {1, 2, 4, 8, 16, 1};
            const int cch[6]  = {128, 128, 128, 96, 64, 32};
            const float* cin2 = ctx_in; int icc = 34;
            float* couts[6] = {pingA, pingB, pingA, pingB, pingA, pingB};
            for (int j = 0; j < 6; ++j) {
                conv(cin2, icc, 0, icc, h, w, ctx_w[j], ctx_b[j],
                     couts[j], cch[j], h, w, 1, dils[j], 1);
                cin2 = couts[j]; icc = cch[j];
            }
            conv(cin2, 32, 0, 32, h, w, ctx_w[6], ctx_b[6], upf, 2, h, w, 1, 1, 0);
            add_kernel<<<blocks1((long)B * 2 * h * w), 256, 0, stream>>>(
                flows[4], upf, (long)B * 2 * h * w);
        }
        flow = flows[i]; ph = h; pw = w; fH[i] = h; fW[i] = w;
    }

    // ---- final 4x upsample of each flow, written reversed into d_out ----
    float* op = (float*)d_out;
    size_t off = 0;
    for (int i = 4; i >= 0; --i) {
        const int h = fH[i], w = fW[i], OH = h * 4, OW = w * 4;
        resize_kernel<<<blocks1((long)B * 2 * OH * OW), 256, 0, stream>>>(
            flows[i], op + off, B, 2, h, w, OH, OW, 4.0f);
        off += (size_t)B * 2 * OH * OW;
    }
}
